// ETransformer_51719996179042
// MI455X (gfx1250) — matrix operand order repaired and checked by reading, not compiled, not run on hardware
//
#include <hip/hip_runtime.h>

typedef __attribute__((ext_vector_type(16))) _Float16 v16h;
typedef __attribute__((ext_vector_type(8)))  float    v8f;

#define DMODEL 128
#define NHEAD  8
#define HDIM   16

// ---------------------------------------------------------------------------
// Zero-fill helper (graph-capture safe; no hipMemsetAsync dependency)
// ---------------------------------------------------------------------------
__global__ void zero_f32(float* __restrict__ p, size_t n) {
    size_t i = (size_t)blockIdx.x * blockDim.x + threadIdx.x;
    if (i < n) p[i] = 0.0f;
}

// ---------------------------------------------------------------------------
// Fused QKV projection: {Q,K,V} = x @ {Wq,Wk,Wv}^T + {bq,bk,bv}
// via v_wmma_f32_16x16x32_f16, sharing the A (x) fragment across all three
// projections: per K-chunk, A is loaded/converted ONCE and fed to 3 WMMAs.
//
// grid = (N/16); block = 256 = 8 waves. Wave w computes three 16x16 output
// tiles (Q/K/V) at rows [16*blockIdx.x,+16), cols [16*w,+16); K loops over
// 4 chunks of 32 -> 12 v_wmma per wave.
//
// Fragment layouts per CDNA5 ISA 7.12.2 (wave32):
//  A (16x32 f16): lane<16 -> row=lane, a[0..7]=K(half*8+0..7), a[8..15]=K(16+half*8+0..7)
//  B (32x16 f16): lane<16 -> col=lane, b[0..7]=K(half*8+0..7), b[8..15]=K(16+half*8+0..7)
//  C/D (16x16 f32): lane<16 -> (m=v, n=lane); lane>=16 -> (m=v+8, n=lane-16)
// ---------------------------------------------------------------------------
__global__ __launch_bounds__(256) void qkv_gemm_wmma(
    const float* __restrict__ x,
    const float* __restrict__ Wq, const float* __restrict__ bq,
    const float* __restrict__ Wk, const float* __restrict__ bk,
    const float* __restrict__ Wv, const float* __restrict__ bv,
    float* __restrict__ Q, float* __restrict__ K, float* __restrict__ V,
    int N)
{
    const float* Ws[3]   = { Wq, Wk, Wv };
    const float* biases[3] = { bq, bk, bv };
    float*       outs[3] = { Q, K, V };

    const int r0   = blockIdx.x * 16;          // node-row tile
    const int wave = threadIdx.x >> 5;         // 0..7
    const int lane = threadIdx.x & 31;
    const int n0   = wave * 16;                // output-column tile
    const int half = lane >> 4;                // 0 or 1 (lane half)
    const int mr   = lane & 15;

    // Clamp A-row reads so EXEC stays all-ones even if N%16 != 0 (N=50000 is exact).
    const int arow_idx = min(r0 + mr, N - 1);

    v8f acc[3] = { v8f{}, v8f{}, v8f{} };

    #pragma unroll
    for (int kc = 0; kc < 4; ++kc) {
        const int k0 = kc * 32;

        // --- A fragment (shared by Q/K/V): x rows, f32 -> f16 in-register ---
        const float* arow = x + (size_t)arow_idx * DMODEL + k0 + half * 8;
        v16h a;
        #pragma unroll
        for (int i = 0; i < 8; ++i) {
            a[i]     = (_Float16)arow[i];        // K = k0 + half*8 + i
            a[i + 8] = (_Float16)arow[16 + i];   // K = k0 + 16 + half*8 + i
        }

        // --- three B fragments + three WMMAs against the same A ---
        #pragma unroll
        for (int m = 0; m < 3; ++m) {
            // B[k][n] = W[n][k]: each lane reads one W row in the A order
            // (b[0..7] = K half*8+0..7, b[8..15] = K 16+half*8+0..7)
            const float* brow = Ws[m] + (size_t)(n0 + mr) * DMODEL + k0 + half * 8;
            v16h b;
            #pragma unroll
            for (int i = 0; i < 16; ++i)
                b[i] = (_Float16)brow[(i < 8) ? i : i + 8];

            acc[m] = __builtin_amdgcn_wmma_f32_16x16x32_f16(
                         /*neg_a=*/false, a, /*neg_b=*/false, b,
                         /*c_mod=*/(short)0, acc[m],
                         /*reuse_a=*/false, /*reuse_b=*/false);
        }
    }

    // --- store D tiles + bias (lanes 0-15 rows 0..7, lanes 16-31 rows 8..15) ---
    const int col   = n0 + mr;
    const int rbase = r0 + (half ? 8 : 0);
    #pragma unroll
    for (int m = 0; m < 3; ++m) {
        const float bb = biases[m][col];
        float* out = outs[m];
        #pragma unroll
        for (int v = 0; v < 8; ++v) {
            int row = rbase + v;
            if (row < N)
                out[(size_t)row * DMODEL + col] = acc[m][v] + bb;
        }
    }
}

// ---------------------------------------------------------------------------
// Edge attention: one wave32 per edge.
// lane = 4*head + w ; each lane owns 4 contiguous elements of its head's
// 16-wide slice -> fully coalesced float4 gathers of Q[dst],K[src],V[src]
// (Q/K/V total 76.8 MB -> resident in the 192 MB L2).
// 4-lane dot reduction via shfl_xor, exp(clamp(dot/4)), then contiguous
// global_atomic_add_f32 scatter into wV (d_out) and Z.
// ---------------------------------------------------------------------------
__global__ __launch_bounds__(256) void edge_attn(
    const float* __restrict__ Q, const float* __restrict__ K,
    const float* __restrict__ V, const int* __restrict__ ei,
    float* __restrict__ out, float* __restrict__ Z, int E)
{
    const int wid = (int)(((size_t)blockIdx.x * blockDim.x + threadIdx.x) >> 5);
    if (wid >= E) return;
    const int lane = threadIdx.x & 31;

    const int src = ei[wid];           // edge_index[0][e]
    const int dst = ei[E + wid];       // edge_index[1][e]

    const int h = lane >> 2;           // head 0..7
    const int w = lane & 3;            // quarter within head

    const size_t qoff = (size_t)dst * DMODEL + h * HDIM + w * 4;
    const size_t koff = (size_t)src * DMODEL + h * HDIM + w * 4;

    const float4 qv = *(const float4*)(Q + qoff);
    const float4 kv = *(const float4*)(K + koff);

    float p = qv.x * kv.x + qv.y * kv.y + qv.z * kv.z + qv.w * kv.w;
    p += __shfl_xor(p, 1, 32);
    p += __shfl_xor(p, 2, 32);         // all 4 lanes of the head hold the dot

    float s = p * 0.25f;               // 1/sqrt(HD), HD=16
    s = __expf(fminf(5.0f, fmaxf(-5.0f, s)));

    const float4 vv = *(const float4*)(V + koff);
    float* op = out + qoff;            // same layout as Q row of dst
    atomicAdd(op + 0, vv.x * s);
    atomicAdd(op + 1, vv.y * s);
    atomicAdd(op + 2, vv.z * s);
    atomicAdd(op + 3, vv.w * s);
    if (w == 0) atomicAdd(Z + (size_t)dst * NHEAD + h, s);
}

// ---------------------------------------------------------------------------
// h_out = wV / (Z + 1e-6)
// ---------------------------------------------------------------------------
__global__ __launch_bounds__(256) void normalize_out(
    float* __restrict__ out, const float* __restrict__ Z, size_t total)
{
    size_t i = (size_t)blockIdx.x * blockDim.x + threadIdx.x;
    if (i >= total) return;
    size_t n = i >> 7;                 // /128
    int    h = (int)((i >> 4) & 7);    // head
    out[i] = out[i] / (Z[n * NHEAD + h] + 1e-6f);
}

// ---------------------------------------------------------------------------
extern "C" void kernel_launch(void* const* d_in, const int* in_sizes, int n_in,
                              void* d_out, int out_size, void* d_ws, size_t ws_size,
                              hipStream_t stream) {
    const float* x   = (const float*)d_in[0];
    const float* Wq  = (const float*)d_in[1];
    const float* bq  = (const float*)d_in[2];
    const float* Wk  = (const float*)d_in[3];
    const float* bk  = (const float*)d_in[4];
    const float* Wv  = (const float*)d_in[5];
    const float* bv  = (const float*)d_in[6];
    const int*   ei  = (const int*)d_in[7];

    const int N = in_sizes[0] / DMODEL;      // 50000
    const int E = in_sizes[7] / 2;           // 800000

    float* out = (float*)d_out;

    // workspace layout (floats): Q | K | V | Z
    float* Qb = (float*)d_ws;
    float* Kb = Qb + (size_t)N * DMODEL;
    float* Vb = Kb + (size_t)N * DMODEL;
    float* Zb = Vb + (size_t)N * DMODEL;

    const size_t outElems = (size_t)N * DMODEL;
    const size_t zElems   = (size_t)N * NHEAD;

    // 1) zero accumulators (out = wV, Zb = Z)
    zero_f32<<<(unsigned)((outElems + 255) / 256), 256, 0, stream>>>(out, outElems);
    zero_f32<<<(unsigned)((zElems   + 255) / 256), 256, 0, stream>>>(Zb, zElems);

    // 2) fused QKV projection (12 WMMAs per wave, shared A fragment)
    qkv_gemm_wmma<<<(N + 15) / 16, 256, 0, stream>>>(x, Wq, bq, Wk, bk, Wv, bv,
                                                     Qb, Kb, Vb, N);

    // 3) per-edge attention + scatter (one wave32 per edge)
    size_t ethreads = (size_t)E * 32;
    edge_attn<<<(unsigned)((ethreads + 255) / 256), 256, 0, stream>>>(
        Qb, Kb, Vb, ei, out, Zb, E);

    // 4) normalize
    normalize_out<<<(unsigned)((outElems + 255) / 256), 256, 0, stream>>>(
        out, Zb, outElems);
}
